// VQVAEBase_33792802685708
// MI455X (gfx1250) — matrix operand order repaired and checked by reading, not compiled, not run on hardware
//
#include <hip/hip_runtime.h>
#include <hip/hip_bf16.h>
#include <math.h>

typedef __attribute__((ext_vector_type(16))) _Float16 v16h;
typedef __attribute__((ext_vector_type(8)))  _Float16 h8;
typedef __attribute__((ext_vector_type(8)))  float    v8f;

#define TM 64
#define TN 128
#define TK 32
#define BPAD 8   // Bs row padding (halves); row stride 40 halves = 80B (16B mult)

// ---------------------------------------------------------------------------
// WMMA fragment layouts (CDNA5 ISA, 16-bit):
// A 16x32: lane L row M=L&15; halves form two contiguous K runs:
//   K = [8*(L>>4) .. +8) and [16+8*(L>>4) .. +8)
// B 32x16: lane L col N=L&15; same two K runs as A: [8*(L>>4) .. +8) and [16+8*(L>>4) .. +8)
// C/D 16x16 f32: VGPR r holds M = r + 8*(L>>4), N = L&15
// ---------------------------------------------------------------------------

// Implicit-im2col fetch; all structural params compile-time so div/mod fold
// and the K-padding guard disappears when KREAL == KPAD.
// MODE 0: regular conv. MODE 1: transposed conv as conv over x2 zero-dilated
// input (PAD here = K-1-pad_orig), flipped/transposed weights pre-packed.
template <int MODE, int KH, int KW, int HIN, int WIN, int HOUT, int WOUT,
          int STRIDE, int PAD, int KREAL, int KPAD>
__device__ __forceinline__ float fetch_patch(const float* __restrict__ In,
                                             int k, int n, int Bn)
{
  if (KREAL != KPAD && k >= KREAL) return 0.0f;
  int kx = k % KW;
  int t  = k / KW;
  int ky = t % KH;
  int ci = t / KH;
  int ox = n % WOUT;
  int t2 = n / WOUT;
  int oy = t2 % HOUT;
  int b  = t2 / HOUT;
  int iy, ix;
  if (MODE == 0) {
    iy = oy * STRIDE - PAD + ky;
    ix = ox * STRIDE - PAD + kx;
    if ((unsigned)iy >= (unsigned)HIN) return 0.0f;
    if ((unsigned)ix >= (unsigned)WIN) return 0.0f;
  } else {
    int dy = oy - PAD + ky;
    int dx = ox - PAD + kx;
    if ((dy | dx) < 0) return 0.0f;
    if ((dy & 1) | (dx & 1)) return 0.0f;
    iy = dy >> 1; ix = dx >> 1;
    if (iy >= HIN || ix >= WIN) return 0.0f;
  }
  const size_t HWin = (size_t)HIN * WIN;
  return In[(size_t)ci * Bn * HWin + (size_t)b * HWin + (size_t)iy * WIN + ix];
}

// Stage one K-chunk: A tile 64x32 (one b128 ld/st per thread) + implicit
// im2col B tile stored transposed [n][k].
template <int MODE, int KH, int KW, int HIN, int WIN, int HOUT, int WOUT,
          int STRIDE, int PAD, int KREAL, int KPAD>
__device__ __forceinline__ void stage_tiles(_Float16 (*__restrict__ Asb)[TK],
                                            _Float16 (*__restrict__ Bsb)[TK + BPAD],
                                            const _Float16* __restrict__ A,
                                            const float* __restrict__ In,
                                            int m0, int n0, int k0, int tid, int Bn)
{
  const int amm = tid >> 2;
  const int akk = (tid & 3) << 3;
  *(h8*)&Asb[amm][akk] = *(const h8*)&A[(size_t)(m0 + amm) * KPAD + (k0 + akk)];

  #pragma unroll
  for (int e = tid; e < TK * TN; e += 256) {
    int kk = e >> 7, nn = e & 127;  // consecutive tid -> consecutive n (coalesced)
    float v = fetch_patch<MODE, KH, KW, HIN, WIN, HOUT, WOUT, STRIDE, PAD,
                          KREAL, KPAD>(In, k0 + kk, n0 + nn, Bn);
    Bsb[nn][kk] = (_Float16)v;
  }
}

// ---------------------------------------------------------------------------
// Generic implicit-GEMM conv:  Out[M][N] = A[M][KPAD] * im2col(In)[KPAD][N] (+bias)
// A pre-packed f16 [Malloc][KPAD], zero padded. 256 threads = 8 waves.
// Macro tile 64(M) x 128(N); wave w owns N sub-tile w, all 4 M sub-tiles.
// Double-buffered LDS: next chunk's global fetch overlaps current WMMAs;
// one barrier per chunk.
// ---------------------------------------------------------------------------
template <int MODE, int KH, int KW, int HIN, int WIN, int HOUT, int WOUT,
          int STRIDE, int PAD, int KREAL, int KPAD>
__global__ __launch_bounds__(256)
void wmma_conv_gemm(const _Float16* __restrict__ A,
                    const float* __restrict__ In,
                    const float* __restrict__ bias,
                    float* __restrict__ Out,
                    int Bn, int Mout, int Nout)
{
  __shared__ __align__(16) _Float16 As[2][TM][TK];         // row = 64B
  __shared__ __align__(16) _Float16 Bs[2][TN][TK + BPAD];  // row = 80B

  const int tid  = threadIdx.x;
  const int lane = tid & 31;
  const int wave = tid >> 5;
  const int n0 = blockIdx.x * TN;
  const int m0 = blockIdx.y * TM;

  v8f acc[4];
  #pragma unroll
  for (int s = 0; s < 4; ++s)
    #pragma unroll
    for (int j = 0; j < 8; ++j) acc[s][j] = 0.0f;

  stage_tiles<MODE, KH, KW, HIN, WIN, HOUT, WOUT, STRIDE, PAD, KREAL, KPAD>(
      As[0], Bs[0], A, In, m0, n0, 0, tid, Bn);
  __syncthreads();

  int p = 0;
  #pragma unroll 1
  for (int k0 = 0; k0 < KPAD; k0 += TK, p ^= 1) {
    // prefetch next chunk into the other buffer (overlaps with WMMAs below)
    if (k0 + TK < KPAD)
      stage_tiles<MODE, KH, KW, HIN, WIN, HOUT, WOUT, STRIDE, PAD, KREAL, KPAD>(
          As[p ^ 1], Bs[p ^ 1], A, In, m0, n0, k0 + TK, tid, Bn);

    // B fragment: two 8-half runs (K=8h.., 16+8h..) -> two aligned ds_load_b128
    const int kb = (lane >> 4) << 3;
    const _Float16* bp = &Bs[p][(wave << 4) + (lane & 15)][kb];
    h8 blo = *(const h8*)bp;
    h8 bhi = *(const h8*)(bp + 16);
    v16h bfrag;
    #pragma unroll
    for (int i = 0; i < 8; ++i) { bfrag[i] = blo[i]; bfrag[i + 8] = bhi[i]; }

    #pragma unroll
    for (int s = 0; s < 4; ++s) {
      // A fragment: two contiguous 8-half runs -> two aligned ds_load_b128
      const _Float16* ap = &As[p][(s << 4) + (lane & 15)][0];
      const int ka = (lane >> 4) << 3;
      h8 alo = *(const h8*)(ap + ka);
      h8 ahi = *(const h8*)(ap + 16 + ka);
      v16h afrag;
      #pragma unroll
      for (int i = 0; i < 8; ++i) { afrag[i] = alo[i]; afrag[i + 8] = ahi[i]; }

      acc[s] = __builtin_amdgcn_wmma_f32_16x16x32_f16(
          false, afrag, false, bfrag, (short)0, acc[s], false, false);
    }
    __syncthreads();  // fills of p^1 done; reads of p done
  }

  const int ncol = n0 + (wave << 4) + (lane & 15);
  #pragma unroll
  for (int s = 0; s < 4; ++s) {
    #pragma unroll
    for (int r = 0; r < 8; ++r) {
      int mrow = m0 + (s << 4) + r + ((lane >> 4) << 3);
      if (mrow < Mout && ncol < Nout) {
        float v = acc[s][r];
        if (bias) v += bias[mrow];
        Out[(size_t)mrow * Nout + ncol] = v;
      }
    }
  }
}

template <int MODE, int KH, int KW, int HIN, int WIN, int HOUT, int WOUT,
          int STRIDE, int PAD, int KREAL, int KPAD>
static void launch_gemm(hipStream_t stream, const _Float16* A, const float* In,
                        const float* bias, float* Out, int Bn, int Mout,
                        int Malloc, int Nout)
{
  dim3 g(Nout / TN, Malloc / TM);
  wmma_conv_gemm<MODE, KH, KW, HIN, WIN, HOUT, WOUT, STRIDE, PAD, KREAL, KPAD>
      <<<g, 256, 0, stream>>>(A, In, bias, Out, Bn, Mout, Nout);
}

// ---------------------------------------------------------------------------
// LayerNorm over channel dim of [C][N] layout (one thread per column n),
// optional skip add + ReLU. Safe in place (thread owns its column).
// ---------------------------------------------------------------------------
__global__ void ln_chan_kernel(float* __restrict__ out, const float* __restrict__ in,
                               const float* __restrict__ g, const float* __restrict__ b,
                               const float* __restrict__ skip, int C, int N, int relu)
{
  int n = blockIdx.x * blockDim.x + threadIdx.x;
  if (n >= N) return;
  float mu = 0.f, m2 = 0.f;
  for (int c = 0; c < C; ++c) {
    float v = in[(size_t)c * N + n];
    mu += v; m2 += v * v;
  }
  mu /= (float)C;
  float var = m2 / (float)C - mu * mu;
  float inv = rsqrtf(var + 1e-5f);
  for (int c = 0; c < C; ++c) {
    float v = (in[(size_t)c * N + n] - mu) * inv * g[c] + b[c];
    if (skip) v += skip[(size_t)c * N + n];
    if (relu) v = fmaxf(v, 0.f);
    out[(size_t)c * N + n] = v;
  }
}

// ---------------------------------------------------------------------------
// Weight packing: f32 -> f16 [Malloc][Kpad], zero padded.
// Regular conv (Cout,Cin,KH,KW) flattens directly to m*Kreal+k.
// ---------------------------------------------------------------------------
__global__ void prep_w_conv(_Float16* __restrict__ A, const float* __restrict__ w,
                            int Cout, int Kreal, int Kpad, int Malloc)
{
  int i = blockIdx.x * blockDim.x + threadIdx.x;
  if (i >= Malloc * Kpad) return;
  int m = i / Kpad, k = i % Kpad;
  float v = 0.f;
  if (m < Cout && k < Kreal) v = w[(size_t)m * Kreal + k];
  A[i] = (_Float16)v;
}

// Transposed conv (Cin,Cout,KH,KW) -> flipped, transposed pack.
__global__ void prep_w_tconv(_Float16* __restrict__ A, const float* __restrict__ w,
                             int Cout, int Cin, int KH, int KW, int Kpad, int Malloc)
{
  int i = blockIdx.x * blockDim.x + threadIdx.x;
  if (i >= Malloc * Kpad) return;
  int m = i / Kpad, k = i % Kpad;
  int Kreal = Cin * KH * KW;
  float v = 0.f;
  if (m < Cout && k < Kreal) {
    int kx = k % KW; int t = k / KW; int ky = t % KH; int ci = t / KH;
    v = w[(((size_t)ci * Cout + m) * KH + (KH - 1 - ky)) * KW + (KW - 1 - kx)];
  }
  A[i] = (_Float16)v;
}

// NCHW x -> normalized [c][b*HW+hw] layout.
__global__ void prep_input_kernel(float* __restrict__ xn, const float* __restrict__ x,
                                  int Bn, int C, int HW)
{
  int i = blockIdx.x * blockDim.x + threadIdx.x;
  if (i >= Bn * C * HW) return;
  int hw = i % HW;
  int c  = (i / HW) % C;
  int b  = i / (HW * C);
  xn[(size_t)c * Bn * HW + (size_t)b * HW + hw] = (x[i] - 128.0f) / 128.0f;
}

__global__ void cnorm_kernel(float* __restrict__ cn, const float* __restrict__ cb,
                             int Kc, int D)
{
  int k = blockIdx.x * blockDim.x + threadIdx.x;
  if (k >= Kc) return;
  float s = 0.f;
  for (int d = 0; d < D; ++d) { float v = cb[(size_t)k * D + d]; s += v * v; }
  cn[k] = s;
}

// argmin_k ( ||c_k||^2 - 2 G[k][n] )  (||e||^2 constant in k)
__global__ void argmin_kernel(int* __restrict__ idx, const float* __restrict__ G,
                              const float* __restrict__ cn, int Kc, int N)
{
  int n = blockIdx.x * blockDim.x + threadIdx.x;
  if (n >= N) return;
  float best = 3.4e38f; int bi = 0;
  for (int k = 0; k < Kc; ++k) {
    float d = cn[k] - 2.0f * G[(size_t)k * N + n];
    if (d < best) { best = d; bi = k; }
  }
  idx[n] = bi;
}

__global__ void gather_kernel(float* __restrict__ emb, const int* __restrict__ idx,
                              const float* __restrict__ cb, int D, int N)
{
  int i = blockIdx.x * blockDim.x + threadIdx.x;
  if (i >= D * N) return;
  int n = i % N, d = i / N;
  emb[i] = cb[(size_t)idx[n] * D + d];
}

// per-batch mean((emb-enc)^2) over [C][b-slice of N]; HW = per-batch spatial
__global__ void vqloss_kernel(float* __restrict__ outl, const float* __restrict__ emb,
                              const float* __restrict__ enc, int C, int HW, int N)
{
  __shared__ float red[256];
  int b = blockIdx.x;
  float s = 0.f;
  for (int e = threadIdx.x; e < C * HW; e += 256) {
    int c = e / HW, hw = e % HW;
    size_t o = (size_t)c * N + (size_t)b * HW + hw;
    float d = emb[o] - enc[o];
    s += d * d;
  }
  red[threadIdx.x] = s;
  __syncthreads();
  for (int st = 128; st > 0; st >>= 1) {
    if (threadIdx.x < st) red[threadIdx.x] += red[threadIdx.x + st];
    __syncthreads();
  }
  if (threadIdx.x == 0) outl[b] = red[0] / (float)(C * HW);
}

// per-batch mean((tanh(pre)-xn)^2)
__global__ void reconloss_kernel(float* __restrict__ outl, const float* __restrict__ pre,
                                 const float* __restrict__ xn, int C, int HW, int N)
{
  __shared__ float red[256];
  int b = blockIdx.x;
  float s = 0.f;
  for (int e = threadIdx.x; e < C * HW; e += 256) {
    int c = e / HW, hw = e % HW;
    size_t o = (size_t)c * N + (size_t)b * HW + hw;
    float d = tanhf(pre[o]) - xn[o];
    s += d * d;
  }
  red[threadIdx.x] = s;
  __syncthreads();
  for (int st = 128; st > 0; st >>= 1) {
    if (threadIdx.x < st) red[threadIdx.x] += red[threadIdx.x + st];
    __syncthreads();
  }
  if (threadIdx.x == 0) outl[b] = red[0] / (float)(C * HW);
}

// out: [loss(16), recon(16), codebook(16), commitment(16)]; commit==codebook fwd
__global__ void final_kernel(float* __restrict__ out, const float* __restrict__ rl,
                             const float* __restrict__ cbl, int Bn)
{
  int b = threadIdx.x;
  if (b >= Bn) return;
  float r = rl[b], c = cbl[b];
  out[b]          = r + c + 0.25f * c;
  out[Bn + b]     = r;
  out[2 * Bn + b] = c;
  out[3 * Bn + b] = c;
}

// ---------------------------------------------------------------------------
extern "C" void kernel_launch(void* const* d_in, const int* in_sizes, int n_in,
                              void* d_out, int out_size, void* d_ws, size_t ws_size,
                              hipStream_t stream)
{
  (void)in_sizes; (void)n_in; (void)out_size; (void)ws_size;
  const int Bn = 16;
  const int N32 = Bn * 32 * 32;     // 16384
  const int N64 = Bn * 64 * 64;     // 65536
  const int N128 = Bn * 128 * 128;  // 262144

  // ---- inputs (setup_inputs dict order, nested dicts flattened in order) ----
  int ii = 0;
  const float* X     = (const float*)d_in[ii++];
  const float* e_c1w = (const float*)d_in[ii++];
  const float* e_c1b = (const float*)d_in[ii++];
  const float* e_n1g = (const float*)d_in[ii++];
  const float* e_n1b = (const float*)d_in[ii++];
  const float* e_c2w = (const float*)d_in[ii++];
  const float* e_c2b = (const float*)d_in[ii++];
  const float* e_n2g = (const float*)d_in[ii++];
  const float* e_n2b = (const float*)d_in[ii++];
  const float* R[4][8]; // e_r1, e_r2, d_r1, d_r2 : c1w,c1b,n1g,n1b,c2w,c2b,n2g,n2b
  for (int r = 0; r < 2; ++r)
    for (int j = 0; j < 8; ++j) R[r][j] = (const float*)d_in[ii++];
  const float* e_c3w = (const float*)d_in[ii++];
  const float* e_c3b = (const float*)d_in[ii++];
  for (int r = 2; r < 4; ++r)
    for (int j = 0; j < 8; ++j) R[r][j] = (const float*)d_in[ii++];
  const float* d_t1w = (const float*)d_in[ii++];
  const float* d_t1b = (const float*)d_in[ii++];
  const float* d_ng  = (const float*)d_in[ii++];
  const float* d_nb  = (const float*)d_in[ii++];
  const float* d_t2w = (const float*)d_in[ii++];
  const float* d_t2b = (const float*)d_in[ii++];
  const float* cbw   = (const float*)d_in[ii++];

  // ---- workspace bump allocator ----
  char* wp = (char*)d_ws;
  auto alloc = [&](size_t bytes) -> void* {
    void* p = (void*)wp;
    wp += (bytes + 255) & ~(size_t)255;
    return p;
  };
  float* xn    = (float*)alloc((size_t)3 * N128 * 4);
  float* buf64 = (float*)alloc((size_t)256 * N64 * 4);
  float* b32a  = (float*)alloc((size_t)256 * N32 * 4);
  float* b32b  = (float*)alloc((size_t)256 * N32 * 4);
  float* b32c  = (float*)alloc((size_t)256 * N32 * 4);
  float* enc   = (float*)alloc((size_t)256 * N32 * 4);
  float* emb   = (float*)alloc((size_t)256 * N32 * 4);
  float* Gm    = (float*)alloc((size_t)128 * N32 * 4);
  int*   idx   = (int*)  alloc((size_t)N32 * 4);
  float* cnrm  = (float*)alloc(128 * 4);
  float* prer  = (float*)alloc((size_t)3 * N128 * 4);
  float* rl    = (float*)alloc(16 * 4);
  float* cbl   = (float*)alloc(16 * 4);
  _Float16* A_ec1 = (_Float16*)alloc((size_t)256 * 64 * 2);
  _Float16* A_ec2 = (_Float16*)alloc((size_t)256 * 4096 * 2);
  _Float16* A_res[8];
  for (int r = 0; r < 8; ++r) A_res[r] = (_Float16*)alloc((size_t)256 * 2304 * 2);
  _Float16* A_ec3 = (_Float16*)alloc((size_t)256 * 256 * 2);
  _Float16* A_cb  = (_Float16*)alloc((size_t)128 * 256 * 2);
  _Float16* A_dt1 = (_Float16*)alloc((size_t)256 * 4096 * 2);
  _Float16* A_dt2 = (_Float16*)alloc((size_t)64 * 4096 * 2);

  auto ln = [&](float* out, const float* in, const float* g_, const float* b_,
                const float* skip, int N, int relu) {
    ln_chan_kernel<<<(N + 255) / 256, 256, 0, stream>>>(out, in, g_, b_, skip, 256, N, relu);
  };
  auto prep = [&](_Float16* A, const float* w, int Cout, int Kreal, int Kpad, int Malloc) {
    prep_w_conv<<<(Malloc * Kpad + 255) / 256, 256, 0, stream>>>(A, w, Cout, Kreal, Kpad, Malloc);
  };

  // ---- weight packing ----
  prep(A_ec1, e_c1w, 256, 48, 64, 256);
  prep(A_ec2, e_c2w, 256, 4096, 4096, 256);
  for (int r = 0; r < 4; ++r) {
    prep(A_res[2 * r],     R[r][0], 256, 2304, 2304, 256);
    prep(A_res[2 * r + 1], R[r][4], 256, 2304, 2304, 256);
  }
  prep(A_ec3, e_c3w, 256, 256, 256, 256);
  prep(A_cb, cbw, 128, 256, 256, 128);
  prep_w_tconv<<<(256 * 4096 + 255) / 256, 256, 0, stream>>>(A_dt1, d_t1w, 256, 256, 4, 4, 4096, 256);
  prep_w_tconv<<<(64 * 4096 + 255) / 256, 256, 0, stream>>>(A_dt2, d_t2w, 3, 256, 4, 4, 4096, 64);

  // ---- input normalize ----
  prep_input_kernel<<<(3 * N128 + 255) / 256, 256, 0, stream>>>(xn, X, Bn, 3, 128 * 128);

  // ---- encoder ----
  launch_gemm<0, 4, 4, 128, 128, 64, 64, 2, 1, 48, 64>(stream, A_ec1, xn, e_c1b,
                                                       buf64, Bn, 256, 256, N64);
  ln(buf64, buf64, e_n1g, e_n1b, nullptr, N64, 1);
  launch_gemm<0, 4, 4, 64, 64, 32, 32, 2, 1, 4096, 4096>(stream, A_ec2, buf64,
                                                         e_c2b, b32a, Bn, 256,
                                                         256, N32);
  ln(b32a, b32a, e_n2g, e_n2b, nullptr, N32, 1);

  float* rot[3] = {b32a, b32b, b32c};
  auto resblock = [&](int r) {
    launch_gemm<0, 3, 3, 32, 32, 32, 32, 1, 1, 2304, 2304>(
        stream, A_res[2 * r], rot[0], R[r][1], rot[1], Bn, 256, 256, N32);
    ln(rot[1], rot[1], R[r][2], R[r][3], nullptr, N32, 1);
    launch_gemm<0, 3, 3, 32, 32, 32, 32, 1, 1, 2304, 2304>(
        stream, A_res[2 * r + 1], rot[1], R[r][5], rot[2], Bn, 256, 256, N32);
    ln(rot[2], rot[2], R[r][6], R[r][7], rot[0], N32, 1);  // +skip, relu
    float* t0 = rot[0]; float* t1 = rot[1];
    rot[0] = rot[2]; rot[1] = t0; rot[2] = t1;
  };
  resblock(0);  // e_r1
  resblock(1);  // e_r2
  launch_gemm<0, 1, 1, 32, 32, 32, 32, 1, 0, 256, 256>(stream, A_ec3, rot[0],
                                                       e_c3b, enc, Bn, 256, 256,
                                                       N32);

  // ---- VQ: G = codebook @ enc ; argmin(||c||^2 - 2G) ; lookup ; losses ----
  launch_gemm<0, 1, 1, 32, 32, 32, 32, 1, 0, 256, 256>(stream, A_cb, enc,
                                                       nullptr, Gm, Bn, 128,
                                                       128, N32);
  cnorm_kernel<<<1, 128, 0, stream>>>(cnrm, cbw, 128, 256);
  argmin_kernel<<<(N32 + 255) / 256, 256, 0, stream>>>(idx, Gm, cnrm, 128, N32);
  gather_kernel<<<(256 * N32 + 255) / 256, 256, 0, stream>>>(emb, idx, cbw, 256, N32);
  vqloss_kernel<<<16, 256, 0, stream>>>(cbl, emb, enc, 256, 32 * 32, N32);

  // ---- decoder (straight-through forward value == emb) ----
  rot[0] = emb; rot[1] = b32a; rot[2] = b32b;
  resblock(2);  // d_r1
  resblock(3);  // d_r2
  launch_gemm<1, 4, 4, 32, 32, 64, 64, 2, 2, 4096, 4096>(stream, A_dt1, rot[0],
                                                         d_t1b, buf64, Bn, 256,
                                                         256, N64);
  ln(buf64, buf64, d_ng, d_nb, nullptr, N64, 1);
  launch_gemm<1, 4, 4, 64, 64, 128, 128, 2, 2, 4096, 4096>(stream, A_dt2, buf64,
                                                           d_t2b, prer, Bn, 3,
                                                           64, N128);

  // ---- losses ----
  reconloss_kernel<<<16, 256, 0, stream>>>(rl, prer, xn, 3, 128 * 128, N128);
  final_kernel<<<1, 32, 0, stream>>>((float*)d_out, rl, cbl, Bn);
}
